// BiLSTMAttention_8787503088026
// MI455X (gfx1250) — matrix operand order repaired and checked by reading, not compiled, not run on hardware
//
#include <hip/hip_runtime.h>
#include <math.h>

// ---------------------------------------------------------------------------
// BiLSTM + attention + MLP for MI455X (gfx1250, wave32, WMMA + async-to-LDS).
//
// Sizes: B=128, T=512, E=300(->pad 320), H=128, 4H=512, 2H=256, FC1=256, C=5.
//
//  1) bf16 conversion of x (K-padded), Wih (K-padded), Whh.
//  2) k_proj : xg[dir][t][b][512] = x@Wih^T + (bih+bhh).
//              A tile (16x320 bf16, shared by all 8 waves) async-DMA'd to LDS
//              once per block (global_load_async_to_lds_b128 / s_wait_asynccnt);
//              v_wmma_f32_16x16x32_bf16 with B fragments streamed from global.
//  3) k_lstm : 16 blocks (2 dirs x 8 batch-slices of 16 rows), 512 steps.
//              Per step: async-DMA the contiguous 32KB xg_t tile into LDS
//              overlapped with the h@Whh^T WMMA loop (acc zero-init), then
//              g_lds += acc, gates in f32, c in registers, h -> LDS bf16.
//  4) k_attn : per-batch softmax attention pooling (f32 VALU).
//  5) k_mlp  : relu(pooled@fc1^T+b) @ fc2^T + b (f32 VALU, tiny).
// ---------------------------------------------------------------------------

#define B_   128
#define T_   512
#define E_   300
#define EP   320   // E padded to multiple of 32 (bf16 WMMA K)
#define H_   128
#define G4   512   // 4*H
#define H2   256   // 2*H
#define FC1_ 256
#define C_   5

typedef __attribute__((ext_vector_type(16))) __bf16       v16bf;
typedef __attribute__((ext_vector_type(8)))  float        v8f;
typedef __attribute__((ext_vector_type(4)))  unsigned int v4u;

union FragU { unsigned int u[8]; v16bf v; };

__device__ __forceinline__ unsigned short f2bf(float f) {
  unsigned int u = __float_as_uint(f);
  unsigned int r = (u + 0x7FFFu + ((u >> 16) & 1u)) >> 16;  // round-nearest-even
  return (unsigned short)r;
}
__device__ __forceinline__ float sigf(float x) { return 1.0f / (1.0f + __expf(-x)); }

// async 16B global->LDS copy (ASYNCcnt-tracked); lds_addr = low 32 bits of
// generic pointer to __shared__ (== LDS byte offset on gfx1250).
__device__ __forceinline__ void async_copy_b128(unsigned lds_addr, const void* gptr) {
  unsigned long long ga = (unsigned long long)(uintptr_t)gptr;
  asm volatile("global_load_async_to_lds_b128 %0, %1, off"
               :: "v"(lds_addr), "v"(ga) : "memory");
}
__device__ __forceinline__ void wait_async0() {
  asm volatile("s_wait_asynccnt 0" ::: "memory");
}

// ---------------------------- bf16 staging ---------------------------------

__global__ void k_conv_x(const float* __restrict__ x, unsigned short* __restrict__ xb) {
  size_t i = (size_t)blockIdx.x * 256 + threadIdx.x;            // B*T*EP
  if (i >= (size_t)B_ * T_ * EP) return;
  size_t row = i / EP; int k = (int)(i - row * EP);
  xb[i] = f2bf(k < E_ ? x[row * (size_t)E_ + k] : 0.0f);
}

__global__ void k_conv_wih(const float* __restrict__ wf, const float* __restrict__ wb,
                           unsigned short* __restrict__ dst) {
  size_t i = (size_t)blockIdx.x * 256 + threadIdx.x;            // 2*G4*EP
  const size_t per = (size_t)G4 * EP;
  int dir = (int)(i / per);
  size_t r = i - (size_t)dir * per;
  size_t row = r / EP; int k = (int)(r - row * EP);
  const float* src = dir ? wb : wf;
  dst[i] = f2bf(k < E_ ? src[row * (size_t)E_ + k] : 0.0f);
}

__global__ void k_conv_whh(const float* __restrict__ wf, const float* __restrict__ wb,
                           unsigned short* __restrict__ dst) {
  size_t i = (size_t)blockIdx.x * 256 + threadIdx.x;            // 2*G4*H
  const size_t per = (size_t)G4 * H_;
  int dir = (int)(i / per);
  size_t r = i - (size_t)dir * per;
  dst[i] = f2bf((dir ? wb : wf)[r]);
}

// ------------------- input projection GEMM (WMMA bf16) ---------------------
// grid: 2 dirs * 4096 row-tiles (t, b0); 8 waves * 4 N-tiles = N=512.

__global__ void __launch_bounds__(256)
k_proj(const unsigned short* __restrict__ xb,   // [B][T][EP] bf16
       const unsigned short* __restrict__ wih,  // [2][G4][EP] bf16
       const float* __restrict__ bih_f, const float* __restrict__ bhh_f,
       const float* __restrict__ bih_b, const float* __restrict__ bhh_b,
       float* __restrict__ xg)                  // [2][T][B][G4] f32
{
  __shared__ __align__(16) unsigned short a_tile[16 * EP];   // 10KB, shared A tile

  const int wave = threadIdx.x >> 5;
  const int lane = threadIdx.x & 31;
  const int lh = lane >> 4, ln = lane & 15;
  const int dir = blockIdx.x >> 12;
  const int rt  = blockIdx.x & 4095;
  const int t   = rt >> 3;
  const int b0  = (rt & 7) << 4;

  // cooperative async DMA of the A tile: 16 rows x 640B (row stride T_*EP in xb)
  for (int i = threadIdx.x; i < 16 * (EP / 8); i += 256) {    // 640 16B chunks
    const int row = i / (EP / 8), col = (i % (EP / 8)) * 8;   // col in elems
    async_copy_b128((unsigned)(uintptr_t)&a_tile[row * EP + col],
                    xb + ((size_t)(b0 + row) * T_ + t) * EP + col);
  }
  wait_async0();
  __syncthreads();

  const float* bih = dir ? bih_b : bih_f;
  const float* bhh = dir ? bhh_b : bhh_f;

  v8f acc[4];
#pragma unroll
  for (int j = 0; j < 4; ++j)
#pragma unroll
    for (int r = 0; r < 8; ++r) acc[j][r] = 0.0f;

  const unsigned short* wbase = wih + (size_t)dir * G4 * EP;

#pragma unroll
  for (int kk = 0; kk < EP; kk += 32) {
    FragU a;
    const int ac = kk + (lh << 3);                 // A: K={0..7,16..23} / {8..15,24..31}
    *(v4u*)&a.u[0] = *(const v4u*)(&a_tile[ln * EP + ac]);
    *(v4u*)&a.u[4] = *(const v4u*)(&a_tile[ln * EP + ac + 16]);
#pragma unroll
    for (int j = 0; j < 4; ++j) {
      const int n = ((wave * 4 + j) << 4) + ln;    // B lane = W row n (contiguous K)
      const unsigned short* brow = wbase + (size_t)n * EP;
      const int bc = kk + (lh << 3);               // B: K={0..7,16..23} / {8..15,24..31}
      FragU b;
      *(v4u*)&b.u[0] = *(const v4u*)(brow + bc);
      *(v4u*)&b.u[4] = *(const v4u*)(brow + bc + 16);
      acc[j] = __builtin_amdgcn_wmma_f32_16x16x32_bf16(false, a.v, false, b.v,
                                                       (short)0, acc[j], false, false);
    }
  }

#pragma unroll
  for (int j = 0; j < 4; ++j) {
    const int n = ((wave * 4 + j) << 4) + ln;
    const float bias = bih[n] + bhh[n];
    float* orow = xg + (((size_t)dir * T_ + t) * B_ + b0) * G4 + n;
#pragma unroll
    for (int r = 0; r < 8; ++r) {
      const int m = r + (lh << 3);                 // D: N=lane%16, M=r(+8)
      orow[(size_t)m * G4] = acc[j][r] + bias;
    }
  }
}

// ------------------- recurrent LSTM (WMMA per timestep) --------------------
// 16 blocks: dir(2) x batch-slice(8 of 16 rows). 8 waves x 4 N-tiles = N=512.
// Per step the 32KB xg tile DMA overlaps the WMMA loop via ASYNCcnt.

__global__ void __launch_bounds__(256)
k_lstm(const float* __restrict__ xg,            // [2][T][B][G4] f32
       const unsigned short* __restrict__ whh,  // [2][G4][H] bf16
       float* __restrict__ out)                 // [B][T][2H] f32
{
  __shared__ __align__(16) float          g_lds[16 * G4];  // 32KB: xg stage + gates
  __shared__ __align__(16) unsigned short h_lds[16 * H_];  //  4KB: h state (bf16)

  const int wave = threadIdx.x >> 5;
  const int lane = threadIdx.x & 31;
  const int lh = lane >> 4, ln = lane & 15;
  const int dir = blockIdx.x >> 3;
  const int b0  = (blockIdx.x & 7) << 4;

  for (int i = threadIdx.x; i < 16 * H_; i += 256) h_lds[i] = 0;
  float c_reg[8];
#pragma unroll
  for (int j = 0; j < 8; ++j) c_reg[j] = 0.0f;
  __syncthreads();

  const unsigned short* wdir = whh + (size_t)dir * G4 * H_;

  for (int s = 0; s < T_; ++s) {
    const int t = dir ? (T_ - 1 - s) : s;

    // (1) async DMA xg_t tile -> g_lds; rows b0..b0+15 are contiguous (32KB).
    {
      const float* gsrc = xg + (((size_t)dir * T_ + t) * B_ + b0) * G4;
#pragma unroll
      for (int kc = 0; kc < 8; ++kc) {                         // 2048 16B chunks
        const int i = threadIdx.x + kc * 256;
        async_copy_b128((unsigned)(uintptr_t)&g_lds[i * 4], gsrc + i * 4);
      }
    }

    // (2) h @ Whh^T (M=16,K=128,N=512), accumulate from zero — overlaps DMA
    v8f acc[4];
#pragma unroll
    for (int j = 0; j < 4; ++j)
#pragma unroll
      for (int r = 0; r < 8; ++r) acc[j][r] = 0.0f;

#pragma unroll
    for (int kk = 0; kk < H_; kk += 32) {
      FragU a;
      const int ac = kk + (lh << 3);
      *(v4u*)&a.u[0] = *(const v4u*)(&h_lds[ln * H_ + ac]);
      *(v4u*)&a.u[4] = *(const v4u*)(&h_lds[ln * H_ + ac + 16]);
#pragma unroll
      for (int j = 0; j < 4; ++j) {
        const int n = ((wave * 4 + j) << 4) + ln;
        const unsigned short* brow = wdir + (size_t)n * H_;
        const int bc = kk + (lh << 3);
        FragU b;
        *(v4u*)&b.u[0] = *(const v4u*)(brow + bc);
        *(v4u*)&b.u[4] = *(const v4u*)(brow + bc + 16);
        acc[j] = __builtin_amdgcn_wmma_f32_16x16x32_bf16(false, a.v, false, b.v,
                                                         (short)0, acc[j], false, false);
      }
    }

    // (3) DMA landed? then g_lds(xg) += WMMA tiles
    wait_async0();
    __syncthreads();
#pragma unroll
    for (int j = 0; j < 4; ++j) {
      const int n = ((wave * 4 + j) << 4) + ln;
#pragma unroll
      for (int r = 0; r < 8; ++r)
        g_lds[(r + (lh << 3)) * G4 + n] += acc[j][r];
    }
    __syncthreads();

    // (4) gate nonlinearity + state update; each thread owns 8 (b,h) cells
    const int base = threadIdx.x * 8;
#pragma unroll
    for (int j = 0; j < 8; ++j) {
      const int id = base + j;
      const int bl = id >> 7, hh = id & 127;
      const float gi = g_lds[bl * G4 + hh];
      const float gf = g_lds[bl * G4 + H_ + hh];
      const float gg = g_lds[bl * G4 + 2 * H_ + hh];
      const float go = g_lds[bl * G4 + 3 * H_ + hh];
      float c = sigf(gf) * c_reg[j] + sigf(gi) * tanhf(gg);
      float h = sigf(go) * tanhf(c);
      c_reg[j] = c;
      h_lds[bl * H_ + hh] = f2bf(h);
      out[((size_t)(b0 + bl) * T_ + t) * H2 + dir * H_ + hh] = h;
    }
    __syncthreads();
  }
}

// ------------------- attention softmax pooling (f32) -----------------------

__global__ void __launch_bounds__(256)
k_attn(const float* __restrict__ lstm_out,      // [B][T][2H]
       const float* __restrict__ attn_w, const float* __restrict__ attn_b,
       float* __restrict__ pooled)              // [B][2H]
{
  __shared__ float aw[H2];
  __shared__ float lg[T_];
  __shared__ float red[256];
  const int b = blockIdx.x, tid = threadIdx.x;
  aw[tid] = attn_w[tid];
  __syncthreads();

  float lmax = -3.0e38f;
  for (int t = tid; t < T_; t += 256) {
    const float* row = lstm_out + ((size_t)b * T_ + t) * H2;
    float sacc = attn_b[0];
    for (int h = 0; h < H2; ++h) sacc += row[h] * aw[h];
    lg[t] = sacc;
    lmax = fmaxf(lmax, sacc);
  }
  red[tid] = lmax; __syncthreads();
  for (int off = 128; off > 0; off >>= 1) {
    if (tid < off) red[tid] = fmaxf(red[tid], red[tid + off]);
    __syncthreads();
  }
  const float mx = red[0]; __syncthreads();

  float lsum = 0.0f;
  for (int t = tid; t < T_; t += 256) {
    float e = __expf(lg[t] - mx);
    lg[t] = e; lsum += e;
  }
  red[tid] = lsum; __syncthreads();
  for (int off = 128; off > 0; off >>= 1) {
    if (tid < off) red[tid] += red[tid + off];
    __syncthreads();
  }
  const float inv = 1.0f / red[0]; __syncthreads();

  float accp = 0.0f;                              // thread tid owns channel tid
  for (int t = 0; t < T_; ++t)
    accp += lg[t] * lstm_out[((size_t)b * T_ + t) * H2 + tid];
  pooled[(size_t)b * H2 + tid] = accp * inv;
}

// ----------------------------- MLP head (f32) ------------------------------

__global__ void __launch_bounds__(256)
k_mlp(const float* __restrict__ pooled,
      const float* __restrict__ fc1_w, const float* __restrict__ fc1_b,
      const float* __restrict__ fc2_w, const float* __restrict__ fc2_b,
      float* __restrict__ y)                      // [B][C]
{
  __shared__ float pl[H2];
  __shared__ float h1[FC1_];
  const int b = blockIdx.x, tid = threadIdx.x;
  pl[tid] = pooled[(size_t)b * H2 + tid];
  __syncthreads();

  float s = fc1_b[tid];
  const float* wrow = fc1_w + (size_t)tid * H2;
  for (int k = 0; k < H2; ++k) s += pl[k] * wrow[k];
  h1[tid] = fmaxf(s, 0.0f);
  __syncthreads();

  if (tid < C_) {
    float o = fc2_b[tid];
    const float* w2 = fc2_w + (size_t)tid * FC1_;
    for (int k = 0; k < FC1_; ++k) o += h1[k] * w2[k];
    y[(size_t)b * C_ + tid] = o;
  }
}

// ------------------------------- launcher ----------------------------------

extern "C" void kernel_launch(void* const* d_in, const int* in_sizes, int n_in,
                              void* d_out, int out_size, void* d_ws, size_t ws_size,
                              hipStream_t stream) {
  const float* x      = (const float*)d_in[0];
  const float* Wih_f  = (const float*)d_in[1];
  const float* Whh_f  = (const float*)d_in[2];
  const float* bih_f  = (const float*)d_in[3];
  const float* bhh_f  = (const float*)d_in[4];
  const float* Wih_b  = (const float*)d_in[5];
  const float* Whh_b  = (const float*)d_in[6];
  const float* bih_b  = (const float*)d_in[7];
  const float* bhh_b  = (const float*)d_in[8];
  const float* attn_w = (const float*)d_in[9];
  const float* attn_b = (const float*)d_in[10];
  const float* fc1_w  = (const float*)d_in[11];
  const float* fc1_b  = (const float*)d_in[12];
  const float* fc2_w  = (const float*)d_in[13];
  const float* fc2_b  = (const float*)d_in[14];
  float* y = (float*)d_out;
  (void)in_sizes; (void)n_in; (void)out_size; (void)ws_size;

  char* ws = (char*)d_ws;
  size_t off = 0;
  auto alloc = [&](size_t bytes) -> void* {
    void* p = ws + off;
    off = (off + bytes + 255) & ~(size_t)255;
    return p;
  };
  unsigned short* xb     = (unsigned short*)alloc((size_t)B_ * T_ * EP * 2);   // 42 MB
  unsigned short* wih    = (unsigned short*)alloc((size_t)2 * G4 * EP * 2);    // 0.7 MB
  unsigned short* whh    = (unsigned short*)alloc((size_t)2 * G4 * H_ * 2);    // 0.3 MB
  float*          xg     = (float*)alloc((size_t)2 * T_ * B_ * G4 * 4);        // 268 MB
  float*          lstm_o = (float*)alloc((size_t)B_ * T_ * H2 * 4);            // 67 MB
  float*          pooled = (float*)alloc((size_t)B_ * H2 * 4);

  { size_t n = (size_t)B_ * T_ * EP;
    k_conv_x<<<dim3((unsigned)((n + 255) / 256)), 256, 0, stream>>>(x, xb); }
  k_conv_wih<<<(2 * G4 * EP) / 256, 256, 0, stream>>>(Wih_f, Wih_b, wih);
  k_conv_whh<<<(2 * G4 * H_) / 256, 256, 0, stream>>>(Whh_f, Whh_b, whh);

  k_proj<<<2 * 4096, 256, 0, stream>>>(xb, wih, bih_f, bhh_f, bih_b, bhh_b, xg);
  k_lstm<<<16, 256, 0, stream>>>(xg, whh, lstm_o);
  k_attn<<<B_, 256, 0, stream>>>(lstm_o, attn_w, attn_b, pooled);
  k_mlp<<<B_, 256, 0, stream>>>(pooled, fc1_w, fc1_b, fc2_w, fc2_b, y);
}
